// DrSpaam_26371099197794
// MI455X (gfx1250) — matrix operand order repaired and checked by reading, not compiled, not run on hardware
//
#include <hip/hip_runtime.h>
#include <hip/hip_bf16.h>

typedef __bf16 bf16_t;
typedef __attribute__((ext_vector_type(16))) bf16_t v16bf;
typedef __attribute__((ext_vector_type(8)))  bf16_t v8bf;
typedef __attribute__((ext_vector_type(8)))  float  v8f;

#define BN_EPS 1e-5f

// ---------------------------------------------------------------------------
// Pack fp32 conv weights (Cout, KK) -> bf16 panel (Cout, Kpad), zero K-tail.
// ---------------------------------------------------------------------------
__global__ void k_packw(const float* __restrict__ w, bf16_t* __restrict__ wp,
                        int KK, int Kpad, int total)
{
    int idx = blockIdx.x * blockDim.x + threadIdx.x;
    if (idx >= total) return;
    int co = idx / Kpad;
    int k  = idx - co * Kpad;
    wp[idx] = (k < KK) ? (bf16_t)w[(size_t)co * KK + k] : (bf16_t)0.0f;
}

// ---------------------------------------------------------------------------
// im2col to bf16 panel, K-major per column: bp[col*Kpad + k], k = ci*KW + dk.
// Handles conv zero-padding, sequence boundaries, pad columns and K-tail.
// ---------------------------------------------------------------------------
__global__ void k_im2col(const float* __restrict__ x, int xoff, int xstr,
                         bf16_t* __restrict__ bp,
                         int colsPad, int totalCols, int Cin, int L, int KW,
                         int pad, int Lout, int Kpad)
{
    int idx = blockIdx.x * blockDim.x + threadIdx.x;
    int total = colsPad * Cin;
    if (idx >= total) return;
    int col = idx / Cin;
    int ci  = idx - col * Cin;
    bf16_t* dst = bp + (size_t)col * Kpad + ci * KW;
    if (col < totalCols) {
        int seq = col / Lout;
        int l   = col - seq * Lout;
        const float* src = x + xoff + (size_t)seq * xstr + ci * L;
        for (int dk = 0; dk < KW; ++dk) {
            int pos = l + dk - pad;
            float v = (pos >= 0 && pos < L) ? src[pos] : 0.0f;
            dst[dk] = (bf16_t)v;
        }
    } else {
        for (int dk = 0; dk < KW; ++dk) dst[dk] = (bf16_t)0.0f;
    }
    if (ci == 0) {
        bf16_t* tail = bp + (size_t)col * Kpad;
        for (int k = Cin * KW; k < Kpad; ++k) tail[k] = (bf16_t)0.0f;
    }
}

// ---------------------------------------------------------------------------
// bf16 WMMA GEMM + bias + BatchNorm(eval) + LeakyReLU(0.1).
// One wave computes a 32x32 D tile: 2 A frags x 2 B frags -> 4 WMMAs / K-step.
// A fragment = two contiguous b128 loads (16-bit A 16x32 layout);
// B fragment = two b128 loads, same k order as A (32x16 B layout).  No divergence in K-loop.
// ---------------------------------------------------------------------------
__global__ __launch_bounds__(256)
void k_gemm_cbl(const bf16_t* __restrict__ wp, const bf16_t* __restrict__ bp,
                const float* __restrict__ cb, const float* __restrict__ bg,
                const float* __restrict__ bbt, const float* __restrict__ bm,
                const float* __restrict__ bv,
                float* __restrict__ out,
                int Kpad, int Cout, int Lout, int totalCols,
                int mtiles, int ntiles)
{
    const int wave = threadIdx.x >> 5;
    const int tile = blockIdx.x * 8 + wave;
    if (tile >= mtiles * ntiles) return;          // wave-uniform guard
    const int mt = tile % mtiles;
    const int nt = tile / mtiles;
    const int lane = threadIdx.x & 31;
    const int kh = lane >> 4;                     // lane half
    const int ln = lane & 15;

    const bf16_t* wr0 = wp + (size_t)(mt * 32 + ln) * Kpad + 8 * kh;
    const bf16_t* wr1 = wr0 + (size_t)16 * Kpad;
    const bf16_t* bc0 = bp + (size_t)(nt * 32 + ln) * Kpad + 8 * kh;
    const bf16_t* bc1 = bc0 + (size_t)16 * Kpad;

    const int ksteps = Kpad >> 5;
    v8f acc00 = {}, acc01 = {}, acc10 = {}, acc11 = {};

    for (int ks = 0; ks < ksteps; ++ks) {
        const int kbase = ks << 5;
        v16bf a0, a1;
        *((v8bf*)&a0)     = *(const v8bf*)(wr0 + kbase);
        *((v8bf*)&a0 + 1) = *(const v8bf*)(wr0 + kbase + 16);
        *((v8bf*)&a1)     = *(const v8bf*)(wr1 + kbase);
        *((v8bf*)&a1 + 1) = *(const v8bf*)(wr1 + kbase + 16);
        v16bf b0, b1;
        *((v8bf*)&b0)     = *(const v8bf*)(bc0 + kbase);
        *((v8bf*)&b0 + 1) = *(const v8bf*)(bc0 + kbase + 16);
        *((v8bf*)&b1)     = *(const v8bf*)(bc1 + kbase);
        *((v8bf*)&b1 + 1) = *(const v8bf*)(bc1 + kbase + 16);
        acc00 = __builtin_amdgcn_wmma_f32_16x16x32_bf16(false, a0, false, b0, (short)0, acc00, false, false);
        acc01 = __builtin_amdgcn_wmma_f32_16x16x32_bf16(false, a0, false, b1, (short)0, acc01, false, false);
        acc10 = __builtin_amdgcn_wmma_f32_16x16x32_bf16(false, a1, false, b0, (short)0, acc10, false, false);
        acc11 = __builtin_amdgcn_wmma_f32_16x16x32_bf16(false, a1, false, b1, (short)0, acc11, false, false);
    }

    auto store = [&](const v8f& acc, int rowBase, int col) {
        if (col < totalCols) {
            int seq = col / Lout;
            int l   = col - seq * Lout;
            #pragma unroll
            for (int v = 0; v < 8; ++v) {
                int c2 = rowBase + v + 8 * kh;    // f32 C/D layout: VGPR v -> M = v + 8*laneHalf
                float y  = acc[v] + cb[c2];
                float sc = bg[c2] * rsqrtf(bv[c2] + BN_EPS);
                y = (y - bm[c2]) * sc + bbt[c2];
                y = (y >= 0.0f) ? y : 0.1f * y;
                out[(size_t)seq * Cout * Lout + c2 * Lout + l] = y;
            }
        }
    };
    const int colA = nt * 32 + ln;
    store(acc00, mt * 32,      colA);
    store(acc01, mt * 32,      colA + 16);
    store(acc10, mt * 32 + 16, colA);
    store(acc11, mt * 32 + 16, colA + 16);
}

// max-pool k=2 s=2 along L.  total = NSEQ*C*Lout
__global__ void k_maxpool2(const float* __restrict__ in, float* __restrict__ out,
                           int total, int Lout)
{
    int idx = blockIdx.x * blockDim.x + threadIdx.x;
    if (idx >= total) return;
    int l  = idx % Lout;
    int sc = idx / Lout;
    const float* p = in + (size_t)sc * (2 * Lout) + 2 * l;
    float a = p[0], b = p[1];
    out[idx] = (a > b) ? a : b;
}

// global average over L=6.  total = NSEQ*128
__global__ void k_avgpool6(const float* __restrict__ in, float* __restrict__ out, int total)
{
    int idx = blockIdx.x * blockDim.x + threadIdx.x;
    if (idx >= total) return;
    const float* p = in + (size_t)idx * 6;
    float s = p[0] + p[1] + p[2] + p[3] + p[4] + p[5];
    out[idx] = s * (1.0f / 6.0f);
}

// windowed attention: dot-128 scores over window<=7, masked softmax,
// memw = sim @ mem, mem_out = 0.5*f4 + 0.5*memw.  One block per (b,n) row.
__global__ __launch_bounds__(256)
void k_attn_update(const float* __restrict__ ex, const float* __restrict__ em,
                   const float* __restrict__ f4, const float* __restrict__ memin,
                   float* __restrict__ memout, float* __restrict__ sim_out,
                   int Nn, int D)
{
    const int bn = blockIdx.x;
    const int b  = bn / Nn;
    const int n  = bn % Nn;
    const int lo = (n - 3 > 0) ? (n - 3) : 0;
    const int hi = (n + 3 < Nn - 1) ? (n + 3) : (Nn - 1);
    const int cnt = hi - lo + 1;

    __shared__ float s_dot[8];
    __shared__ float s_sims[8];

    const int tid = threadIdx.x;
    if (tid < cnt) {
        const float* xe = ex + (size_t)(b * Nn + n) * 128;
        const float* me = em + (size_t)(b * Nn + lo + tid) * 128;
        float s = 0.0f;
        #pragma unroll 4
        for (int k = 0; k < 128; ++k) s += xe[k] * me[k];
        s_dot[tid] = s;
    }
    __syncthreads();
    if (tid == 0) {
        float mx = s_dot[0];
        for (int j = 1; j < cnt; ++j) mx = (s_dot[j] > mx) ? s_dot[j] : mx;
        float sum = 0.0f;
        for (int j = 0; j < cnt; ++j) { float e = __expf(s_dot[j] - mx); s_sims[j] = e; sum += e; }
        float inv = 1.0f / sum;
        for (int j = 0; j < cnt; ++j) s_sims[j] *= inv;
    }
    __syncthreads();
    if (tid < cnt)
        sim_out[(size_t)(b * Nn + n) * Nn + lo + tid] = s_sims[tid];

    const size_t rowOut = (size_t)(b * Nn + n) * D;
    for (int d = tid; d < D; d += blockDim.x) {
        float acc = 0.0f;
        for (int j = 0; j < cnt; ++j)
            acc += s_sims[j] * memin[(size_t)(b * Nn + lo + j) * D + d];
        memout[rowOut + d] = 0.5f * f4[rowOut + d] + 0.5f * acc;
    }
}

// 1x1 heads: cls (1 ch) + reg (2 ch) from 128-d pooled features
__global__ void k_head(const float* __restrict__ flow,
                       const float* __restrict__ cls_w, const float* __restrict__ cls_b,
                       const float* __restrict__ reg_w, const float* __restrict__ reg_b,
                       float* __restrict__ out_cls, float* __restrict__ out_reg, int NSEQ)
{
    int s = blockIdx.x * blockDim.x + threadIdx.x;
    if (s >= NSEQ) return;
    const float* f = flow + (size_t)s * 128;
    float c = 0.0f, r0 = 0.0f, r1 = 0.0f;
    #pragma unroll 4
    for (int k = 0; k < 128; ++k) {
        float fv = f[k];
        c  += fv * cls_w[k];
        r0 += fv * reg_w[k];
        r1 += fv * reg_w[128 + k];
    }
    out_cls[s]         = c  + cls_b[0];
    out_reg[2 * s]     = r0 + reg_b[0];
    out_reg[2 * s + 1] = r1 + reg_b[1];
}

__global__ void k_fill0(float* __restrict__ p, int n)
{
    int idx = blockIdx.x * blockDim.x + threadIdx.x;
    if (idx < n) p[idx] = 0.0f;
}

// ---------------------------------------------------------------------------
// Host orchestration
// ---------------------------------------------------------------------------
struct CLayer { const float *w, *b, *g, *bt, *m, *v; };

static void launch_conv(hipStream_t s, const float* x, int xoff, int xstr,
                        const CLayer& L, float* out, bf16_t* wpk, bf16_t* bpk,
                        int NSEQ, int Cin, int Lin, int KW, int pad,
                        int Cout, int Lout)
{
    int KK   = Cin * KW;
    int Kpad = (KK + 31) & ~31;
    int totalCols = NSEQ * Lout;
    int colsPad   = (totalCols + 31) & ~31;
    {
        int tot = Cout * Kpad;
        k_packw<<<(tot + 255) / 256, 256, 0, s>>>(L.w, wpk, KK, Kpad, tot);
    }
    {
        int tot = colsPad * Cin;
        k_im2col<<<(tot + 255) / 256, 256, 0, s>>>(x, xoff, xstr, bpk, colsPad,
                                                   totalCols, Cin, Lin, KW, pad,
                                                   Lout, Kpad);
    }
    int mtiles = Cout / 32;
    int ntiles = colsPad / 32;
    int tiles  = mtiles * ntiles;
    k_gemm_cbl<<<(tiles + 7) / 8, 256, 0, s>>>(wpk, bpk, L.b, L.g, L.bt, L.m, L.v,
                                               out, Kpad, Cout, Lout, totalCols,
                                               mtiles, ntiles);
}

extern "C" void kernel_launch(void* const* d_in, const int* in_sizes, int n_in,
                              void* d_out, int out_size, void* d_ws, size_t ws_size,
                              hipStream_t stream)
{
    const int Bb = 4, Nn = 450, Ss = 5, Pp = 48;
    const int NSEQ = Bb * Nn;                       // 1800
    const int SIG_ELEMS = Bb * Nn * Ss * Pp;        // 432000

    // ---- locate signal + param leaves (robust to pytree flatten order) ----
    int sigIdx = -1;
    for (int i = 0; i < n_in; ++i)
        if (in_sizes[i] == SIG_ELEMS) { sigIdx = i; break; }
    if (sigIdx < 0) sigIdx = 0;
    const float* signal = (const float*)d_in[sigIdx];

    const float* P[96]; int sz[96]; int np = 0;
    for (int i = 0; i < n_in && np < 96; ++i)
        if (i != sigIdx) { P[np] = (const float*)d_in[i]; sz[np] = in_sizes[i]; ++np; }
    if (np < 76) return;

    bool topSorted, dictSorted;
    if      (sz[0] == 128)    { topSorted = true;  dictSorted = true;  } // attn.b
    else if (sz[0] == 393216) { topSorted = true;  dictSorted = false; } // attn.w
    else if (sz[0] == 64)     { topSorted = false; dictSorted = true;  } // u1[0].b
    else                      { topSorted = false; dictSorted = false; } // u1[0].w (192)

    auto getLayer = [&](int base) -> CLayer {
        CLayer L;
        if (dictSorted) { // b, bt, g, m, v, w
            L.b = P[base]; L.bt = P[base + 1]; L.g = P[base + 2];
            L.m = P[base + 3]; L.v = P[base + 4]; L.w = P[base + 5];
        } else {          // w, b, g, bt, m, v
            L.w = P[base]; L.b = P[base + 1]; L.g = P[base + 2];
            L.bt = P[base + 3]; L.m = P[base + 4]; L.v = P[base + 5];
        }
        return L;
    };

    CLayer u1[3], u2[3], u3[3], u4[2], attn;
    const float *cls_w, *cls_b, *reg_w, *reg_b;
    if (topSorted) {  // attn, cls_b, cls_w, reg_b, reg_w, u1, u2, u3, u4
        attn  = getLayer(0);
        cls_b = P[6]; cls_w = P[7]; reg_b = P[8]; reg_w = P[9];
        int o = 10;
        for (int j = 0; j < 3; ++j) u1[j] = getLayer(o + 6 * j); o += 18;
        for (int j = 0; j < 3; ++j) u2[j] = getLayer(o + 6 * j); o += 18;
        for (int j = 0; j < 3; ++j) u3[j] = getLayer(o + 6 * j); o += 18;
        u4[0] = getLayer(o); u4[1] = getLayer(o + 6);
    } else {          // u1, u2, u3, u4, attn, cls_w, cls_b, reg_w, reg_b
        for (int j = 0; j < 3; ++j) u1[j] = getLayer(6 * j);
        for (int j = 0; j < 3; ++j) u2[j] = getLayer(18 + 6 * j);
        for (int j = 0; j < 3; ++j) u3[j] = getLayer(36 + 6 * j);
        u4[0] = getLayer(54); u4[1] = getLayer(60);
        attn  = getLayer(66);
        cls_w = P[72]; cls_b = P[73]; reg_w = P[74]; reg_b = P[75];
    }

    // ---- workspace layout (float units; all offsets multiples of 8) ----
    float* ws = (float*)d_ws;
    size_t o = 0;
    const size_t BIG = (size_t)NSEQ * 128 * 48;     // 11,059,200 (= 1800*512*12)
    const size_t MEM = (size_t)NSEQ * 256 * 12;     // 5,529,600
    float* bufA = ws + o; o += BIG;
    float* bufB = ws + o; o += BIG;
    float* memA = ws + o; o += MEM;
    float* memB = ws + o; o += MEM;
    float* f4b  = ws + o; o += MEM;
    float* exb  = ws + o; o += (size_t)NSEQ * 128;
    float* emb  = ws + o; o += (size_t)NSEQ * 128;
    bf16_t* wpk = (bf16_t*)(ws + o); o += 393216;        // 786,432 bf16 capacity
    bf16_t* bpk = (bf16_t*)(ws + o); o += 8294400;       // 16,588,800 bf16 capacity
    (void)ws_size;

    float* out_cls = (float*)d_out;                 // 1800
    float* out_reg = out_cls + NSEQ;                // 3600
    float* out_sim = out_cls + NSEQ * 3;            // 810000
    (void)out_size;

    // zero sim once (final scan's window values overwrite the same positions)
    {
        int n = Bb * Nn * Nn;
        k_fill0<<<(n + 255) / 256, 256, 0, stream>>>(out_sim, n);
    }

    float* memCur = memA;
    float* memOth = memB;

    for (int i = 0; i < Ss; ++i) {
        // ---- u1: (BN,1,48) -> (BN,128,24) ----
        launch_conv(stream, signal, i * Pp, Ss * Pp, u1[0], bufA, wpk, bpk, NSEQ, 1,  48, 3, 1, 64,  48);
        launch_conv(stream, bufA,   0, 64 * 48,      u1[1], bufB, wpk, bpk, NSEQ, 64, 48, 3, 1, 64,  48);
        launch_conv(stream, bufB,   0, 64 * 48,      u1[2], bufA, wpk, bpk, NSEQ, 64, 48, 3, 1, 128, 48);
        {
            int tot = NSEQ * 128 * 24;
            k_maxpool2<<<(tot + 255) / 256, 256, 0, stream>>>(bufA, bufB, tot, 24);
        }
        // ---- u2: (BN,128,24) -> (BN,256,12) ----
        launch_conv(stream, bufB, 0, 128 * 24, u2[0], bufA, wpk, bpk, NSEQ, 128, 24, 3, 1, 128, 24);
        launch_conv(stream, bufA, 0, 128 * 24, u2[1], bufB, wpk, bpk, NSEQ, 128, 24, 3, 1, 128, 24);
        launch_conv(stream, bufB, 0, 128 * 24, u2[2], bufA, wpk, bpk, NSEQ, 128, 24, 3, 1, 256, 24);
        float* f4dst = (i == 0) ? memCur : f4b;
        {
            int tot = NSEQ * 256 * 12;
            k_maxpool2<<<(tot + 255) / 256, 256, 0, stream>>>(bufA, f4dst, tot, 12);
        }
        // ---- recurrent memory update (scans 1..4) ----
        if (i > 0) {
            launch_conv(stream, f4b,    0, 256 * 12, attn, exb, wpk, bpk, NSEQ, 256, 12, 12, 0, 128, 1);
            launch_conv(stream, memCur, 0, 256 * 12, attn, emb, wpk, bpk, NSEQ, 256, 12, 12, 0, 128, 1);
            k_attn_update<<<Bb * Nn, 256, 0, stream>>>(exb, emb, f4b, memCur, memOth,
                                                       out_sim, Nn, 256 * 12);
            float* t = memCur; memCur = memOth; memOth = t;
        }
    }

    // ---- u3: (BN,256,12) -> (BN,512,6) ----
    launch_conv(stream, memCur, 0, 256 * 12, u3[0], bufA, wpk, bpk, NSEQ, 256, 12, 3, 1, 256, 12);
    launch_conv(stream, bufA,   0, 256 * 12, u3[1], bufB, wpk, bpk, NSEQ, 256, 12, 3, 1, 256, 12);
    launch_conv(stream, bufB,   0, 256 * 12, u3[2], bufA, wpk, bpk, NSEQ, 256, 12, 3, 1, 512, 12);
    {
        int tot = NSEQ * 512 * 6;
        k_maxpool2<<<(tot + 255) / 256, 256, 0, stream>>>(bufA, bufB, tot, 6);
    }
    // ---- u4: (BN,512,6) -> (BN,128,6) ----
    launch_conv(stream, bufB, 0, 512 * 6, u4[0], bufA, wpk, bpk, NSEQ, 512, 6, 3, 1, 256, 6);
    launch_conv(stream, bufA, 0, 256 * 6, u4[1], bufB, wpk, bpk, NSEQ, 256, 6, 3, 1, 128, 6);

    // ---- global average pool + heads ----
    {
        int tot = NSEQ * 128;
        k_avgpool6<<<(tot + 255) / 256, 256, 0, stream>>>(bufB, bufA, tot);
    }
    k_head<<<(NSEQ + 255) / 256, 256, 0, stream>>>(bufA, cls_w, cls_b, reg_w, reg_b,
                                                   out_cls, out_reg, NSEQ);
}
